// IntegratedModel_35588099014947
// MI455X (gfx1250) — matrix operand order repaired and checked by reading, not compiled, not run on hardware
//
#include <hip/hip_runtime.h>
#include <hip/hip_bf16.h>

// ---------------------------------------------------------------------------
// MI455X (gfx1250) implementation.
//   B=64, L=2048, H=256, V=32000, HALF=128
// K0: f32 -> bf16 weight conversion (w1, w2, semW, epiW, outW)
// K1: fused embed-gather + FFN (bf16 WMMA) + residual + LayerNorm + k-proj
// K2: sequential associative scan, M_s/M_e resident in LDS (one WG / batch),
//     per-step k-vectors double-buffered via GLOBAL_LOAD_ASYNC_TO_LDS (+ s_wait_asynccnt)
// K3: readout (q projections, M readback, rp projection)
// K4: output GEMM 64x32000 via bf16 WMMA
// ---------------------------------------------------------------------------

typedef __bf16 bf16_t;
typedef bf16_t v16bf __attribute__((ext_vector_type(16)));
typedef float  v8f   __attribute__((ext_vector_type(8)));
typedef int    v4i_b __attribute__((vector_size(4 * sizeof(int))));

static_assert(sizeof(bf16_t) == 2, "bf16 size");
static_assert(sizeof(v16bf) == 32, "v16bf size");

#if defined(__AMDGCN__) &&                                                    \
    __has_builtin(__builtin_amdgcn_global_load_async_to_lds_b128) &&          \
    __has_builtin(__builtin_amdgcn_s_wait_asynccnt)
#define USE_ASYNC_LDS 1
#else
#define USE_ASYNC_LDS 0
#endif

union FragBF { uint4 u4[2]; v16bf v; };

// A fragment (16x32 bf16, M x K), row-major source with leading dim lda.
// ISA layout: lane m=lane&15, half=lane>>4; elements 0..7 -> K = 8*half..8*half+7,
// elements 8..15 -> K = 16+8*half .. 23+8*half  (two contiguous 16B runs).
__device__ __forceinline__ v16bf frag_a(const bf16_t* base, int lda, int lane) {
  const int m = lane & 15, h = lane >> 4;
  FragBF f;
  const bf16_t* p = base + m * lda + 8 * h;
  f.u4[0] = *(const uint4*)(p);
  f.u4[1] = *(const uint4*)(p + 16);
  return f.v;
}

// B fragment (32x16 bf16, K x N) sourced from weight stored as W[n][k] row-major
// (ldb = K stride). Lane n=lane&15, half=lane>>4; elements 0..7 -> K = 8*half..8*half+7,
// elements 8..15 -> K = 16+8*half .. 23+8*half  (two contiguous 16B runs).
__device__ __forceinline__ v16bf frag_b(const bf16_t* base, int ldb, int lane) {
  const int n = lane & 15, h = lane >> 4;
  FragBF f;
  const bf16_t* p = base + n * ldb + 8 * h;
  f.u4[0] = *(const uint4*)(p);
  f.u4[1] = *(const uint4*)(p + 16);
  return f.v;
}

__device__ __forceinline__ v8f wmma_bf16(v16bf a, v16bf b, v8f c) {
  return __builtin_amdgcn_wmma_f32_16x16x32_bf16(false, a, false, b, (short)0, c,
                                                 false, false);
}

__device__ __forceinline__ v8f vzero8() {
  v8f z = {0.f, 0.f, 0.f, 0.f, 0.f, 0.f, 0.f, 0.f};
  return z;
}

// ---------------------------------------------------------------------------
// K0: f32 -> bf16 conversion
// ---------------------------------------------------------------------------
__global__ void cvt_bf16_kernel(const float* __restrict__ src,
                                bf16_t* __restrict__ dst, int n) {
  int i = blockIdx.x * blockDim.x + threadIdx.x;
  int stride = gridDim.x * blockDim.x;
  for (; i < n; i += stride) dst[i] = (bf16_t)src[i];
}

// ---------------------------------------------------------------------------
// K1: fused token pipeline. 128 threads = 4 waves, wave owns 16 tokens.
//   h0 = embed_W[seq]                     (gather, bf16 into LDS)
//   f  = relu(h0 @ w1^T + b1)             (WMMA 16x512)
//   x  = f @ w2^T + b2 + h0               (WMMA, residual in place)
//   h  = LN(x)*g + b                      (bf16 in place, f32 q for l==L-1)
//   ks = h @ semW^T ; ke = h @ epiW^T     (WMMA, -> ks_all/ke_all [l][b][k])
// ---------------------------------------------------------------------------
__global__ void __launch_bounds__(128) token_fused_kernel(
    const int* __restrict__ seq, const float* __restrict__ embed_W,
    const bf16_t* __restrict__ w1, const float* __restrict__ b1,
    const bf16_t* __restrict__ w2, const float* __restrict__ b2,
    const float* __restrict__ ln_g, const float* __restrict__ ln_b,
    const bf16_t* __restrict__ semW, const bf16_t* __restrict__ epiW,
    float* __restrict__ ks_all, float* __restrict__ ke_all,
    float* __restrict__ qout) {
  __shared__ bf16_t Abuf[4][16][256];  // 32 KB : h0 -> x -> h
  __shared__ bf16_t Fbuf[4][16][512];  // 64 KB : relu activations
  __shared__ float red1[128], red2[128], smu[64], srs[64];

  const int tid = threadIdx.x;
  const int w = tid >> 5, lane = tid & 31;
  const int blk_tok0 = blockIdx.x * 64;
  const int wav_tok0 = blk_tok0 + w * 16;

  // ---- embedding gather -> Abuf (bf16) ----
  for (int t = 0; t < 16; ++t) {
    const int tok = wav_tok0 + t;
    const float* src = embed_W + (size_t)seq[tok] * 256;
#pragma unroll
    for (int c = 0; c < 8; ++c) {
      const int col = lane + 32 * c;
      Abuf[w][t][col] = (bf16_t)src[col];
    }
  }
  __syncthreads();

  // ---- GEMM1: F = relu(A @ w1^T + b1), 16x256 * 256x512 ----
  for (int nt = 0; nt < 32; ++nt) {
    v8f acc = vzero8();
#pragma unroll
    for (int kt = 0; kt < 8; ++kt) {
      v16bf a = frag_a(&Abuf[w][0][kt * 32], 256, lane);
      v16bf bb = frag_b(w1 + (size_t)(nt * 16) * 256 + kt * 32, 256, lane);
      acc = wmma_bf16(a, bb, acc);
    }
    const int n = lane & 15, hf = lane >> 4;
    const float bias = b1[nt * 16 + n];
#pragma unroll
    for (int r = 0; r < 8; ++r) {
      const int m = r + 8 * hf;
      float v = acc[r] + bias;
      Fbuf[w][m][nt * 16 + n] = (bf16_t)(v > 0.f ? v : 0.f);
    }
  }
  __syncthreads();

  // ---- GEMM2 + residual (in place into Abuf): x = F @ w2^T + b2 + h0 ----
  for (int nt = 0; nt < 16; ++nt) {
    v8f acc = vzero8();
#pragma unroll
    for (int kt = 0; kt < 16; ++kt) {
      v16bf a = frag_a(&Fbuf[w][0][kt * 32], 512, lane);
      v16bf bb = frag_b(w2 + (size_t)(nt * 16) * 512 + kt * 32, 512, lane);
      acc = wmma_bf16(a, bb, acc);
    }
    const int n = lane & 15, hf = lane >> 4;
    const float bias = b2[nt * 16 + n];
#pragma unroll
    for (int r = 0; r < 8; ++r) {
      const int m = r + 8 * hf;
      const int col = nt * 16 + n;
      float x = acc[r] + bias + (float)Abuf[w][m][col];
      Abuf[w][m][col] = (bf16_t)x;
    }
  }
  __syncthreads();

  // ---- LayerNorm per token (2 threads / token) ----
  {
    const int token = tid >> 1;  // 0..63 within block
    const int sub = tid & 1;
    const int tw = token >> 4, tt = token & 15;
    float s = 0.f, s2 = 0.f;
    for (int c = 0; c < 128; ++c) {
      float v = (float)Abuf[tw][tt][sub * 128 + c];
      s += v;
      s2 += v * v;
    }
    red1[tid] = s;
    red2[tid] = s2;
    __syncthreads();
    if (sub == 0) {
      float S = red1[tid] + red1[tid + 1];
      float S2 = red2[tid] + red2[tid + 1];
      float mu = S * (1.f / 256.f);
      float var = S2 * (1.f / 256.f) - mu * mu;
      smu[token] = mu;
      srs[token] = rsqrtf(var + 1e-5f);
    }
    __syncthreads();
    const float mu = smu[token], rstd = srs[token];
    const int tok = blk_tok0 + token;
    const int l = tok & 2047;  // L = 2048
    const int bidx = tok >> 11;
    const bool isq = (l == 2047);
    for (int c = 0; c < 128; ++c) {
      const int col = sub * 128 + c;
      float v = (float)Abuf[tw][tt][col];
      float hh = (v - mu) * rstd * ln_g[col] + ln_b[col];
      Abuf[tw][tt][col] = (bf16_t)hh;
      if (isq) qout[bidx * 256 + col] = hh;
    }
  }
  __syncthreads();

  // ---- k projections: h @ {semW,epiW}^T -> ks_all/ke_all [l][b][128] ----
  for (int which = 0; which < 2; ++which) {
    const bf16_t* Wk = which ? epiW : semW;
    float* dst = which ? ke_all : ks_all;
    for (int nt = 0; nt < 8; ++nt) {
      v8f acc = vzero8();
#pragma unroll
      for (int kt = 0; kt < 8; ++kt) {
        v16bf a = frag_a(&Abuf[w][0][kt * 32], 256, lane);
        v16bf bb = frag_b(Wk + (size_t)(nt * 16) * 256 + kt * 32, 256, lane);
        acc = wmma_bf16(a, bb, acc);
      }
      const int n = lane & 15, hf = lane >> 4;
#pragma unroll
      for (int r = 0; r < 8; ++r) {
        const int m = r + 8 * hf;
        const int tok = wav_tok0 + m;
        const int l = tok & 2047;
        const int bidx = tok >> 11;
        if (l < 2047)
          dst[((size_t)l * 64 + bidx) * 128 + nt * 16 + n] = acc[r];
      }
    }
  }
}

// ---------------------------------------------------------------------------
// K2: sequential scan. One workgroup per batch; M_s/M_e live in LDS with
// bank-conflict padding (pitch 129). Rank-1 algebra: energy uses
// mean(D*D) = |r|^2 * |kn|^2 / 16384, update is M += 0.05*gate * r (x) kn.
// Threads 0..127 own M_s rows, 128..255 own M_e rows. The next step's
// k-vectors are double-buffered into LDS with async loads (ASYNCcnt) so the
// global-memory latency overlaps the matvec / rank-1 update of this step.
// ---------------------------------------------------------------------------
__global__ void __launch_bounds__(256) scan_kernel(
    const float* __restrict__ ks_all, const float* __restrict__ ke_all,
    float* __restrict__ Ms_out, float* __restrict__ Me_out) {
  __shared__ float Ms[128 * 129];  // 66048 B
  __shared__ float Me[128 * 129];  // 66048 B
  __shared__ float kbuf[2][256];   // [buf][ ks(0..127) | ke(128..255) ]
  __shared__ float kns[128], kne[128], rs[128], re[128];
  __shared__ float red[256];
  __shared__ float s_scal[4];  // inv_s, inv_e, |kns|^2, |kne|^2
  __shared__ int s_gate;

  const int tid = threadIdx.x;
  const int b = blockIdx.x;
  const int half = tid >> 7;  // 0 = semantic, 1 = episodic
  const int i = tid & 127;

#if USE_ASYNC_LDS
  // issue async load of step 0's k-vectors while we zero M.
  // 64 lanes x b128 = 256 floats (ks || ke).
  if (tid < 64) {
    const int part = tid >> 5;       // 0 = ks, 1 = ke
    const int idx = (tid & 31) * 4;  // float index within the 128-vector
    const float* src =
        (part ? ke_all : ks_all) + ((size_t)0 * 64 + b) * 128 + idx;
    __builtin_amdgcn_global_load_async_to_lds_b128(
        (v4i_b*)(src),
        (__attribute__((address_space(3))) v4i_b*)(&kbuf[0][part * 128 + idx]),
        0, 0);
  }
#endif

  for (int k = tid; k < 128 * 129; k += 256) {
    Ms[k] = 0.f;
    Me[k] = 0.f;
  }
#if USE_ASYNC_LDS
  __builtin_amdgcn_s_wait_asynccnt(0);
#endif
  __syncthreads();

  for (int l = 0; l < 2047; ++l) {
    const int cur = l & 1;
#if USE_ASYNC_LDS
    // kick off next step's k-vectors into the other buffer (consumed in step l+1,
    // after the step-boundary barrier; buffer (l+1)&1 was last read in step l-1).
    if (l + 1 < 2047 && tid < 64) {
      const int part = tid >> 5;
      const int idx = (tid & 31) * 4;
      const float* src =
          (part ? ke_all : ks_all) + ((size_t)(l + 1) * 64 + b) * 128 + idx;
      __builtin_amdgcn_global_load_async_to_lds_b128(
          (v4i_b*)(src),
          (__attribute__((address_space(3))) v4i_b*)(
              &kbuf[(l + 1) & 1][part * 128 + idx]),
          0, 0);
    }
#else
    {
      const size_t base = ((size_t)l * 64 + b) * 128;
      kbuf[cur][tid] = half ? ke_all[base + i] : ks_all[base + i];
      if (l + 1 < 2047) {
        const size_t nb = ((size_t)(l + 1) * 64 + b) * 128;
        __builtin_prefetch(half ? &ke_all[nb + i] : &ks_all[nb + i], 0, 0);
      }
    }
    __syncthreads();
#endif
    const float kval = kbuf[cur][tid];
    red[tid] = kval * kval;
    __syncthreads();
    for (int s = 64; s > 0; s >>= 1) {
      if (i < s) red[tid] += red[tid + s];
      __syncthreads();
    }
    if (tid == 0) {
      float nrm = sqrtf(red[0]);
      float den = fmaxf(nrm, 1e-12f);
      s_scal[0] = 1.f / den;
      s_scal[2] = red[0] / (den * den);
    } else if (tid == 128) {
      float nrm = sqrtf(red[128]);
      float den = fmaxf(nrm, 1e-12f);
      s_scal[1] = 1.f / den;
      s_scal[3] = red[128] / (den * den);
    }
    __syncthreads();
    if (half) kne[i] = kval * s_scal[1];
    else      kns[i] = kval * s_scal[0];
    __syncthreads();

    // matvec: r_i = k_i - sum_j M[i][j]*kn[j]
    float rval;
    {
      const float* row = (half ? Me : Ms) + i * 129;
      const float* kn = half ? kne : kns;
      float acc = 0.f;
#pragma unroll 8
      for (int j = 0; j < 128; ++j) acc = fmaf(row[j], kn[j], acc);
      rval = kval - acc;
      if (half) re[i] = rval; else rs[i] = rval;
      red[tid] = rval * rval;
    }
    __syncthreads();
    for (int s = 64; s > 0; s >>= 1) {
      if (i < s) red[tid] += red[tid + s];
      __syncthreads();
    }
    if (tid == 0) {
      float energy =
          0.5f * (red[0] * s_scal[2] + red[128] * s_scal[3]) * (1.f / 16384.f);
      s_gate = (energy >= 0.4f) ? 1 : 0;
    }
    __syncthreads();
    if (s_gate) {
      float* row = (half ? Me : Ms) + i * 129;
      const float* kn = half ? kne : kns;
      const float c = 0.05f * rval;
#pragma unroll 8
      for (int j = 0; j < 128; ++j) row[j] = fmaf(c, kn[j], row[j]);
    }
#if USE_ASYNC_LDS
    // retire the async load for step l+1 before the step boundary barrier.
    __builtin_amdgcn_s_wait_asynccnt(0);
#endif
    __syncthreads();
  }

  // write final state
  {
    float* dst = (half ? Me_out : Ms_out) + (size_t)b * 16384 + (size_t)i * 128;
    const float* row = (half ? Me : Ms) + i * 129;
    for (int j = 0; j < 128; ++j) dst[j] = row[j];
  }
}

// ---------------------------------------------------------------------------
// K3: readout. qs/qe = q @ {semW,epiW}^T ; cs/ce = M @ qs/qe ;
//     rb = [cs,ce] @ rpW^T + rp_b   (f32 + bf16 copies)
// ---------------------------------------------------------------------------
__global__ void __launch_bounds__(256) readout_kernel(
    const float* __restrict__ qout, const float* __restrict__ semW,
    const float* __restrict__ epiW, const float* __restrict__ Ms_g,
    const float* __restrict__ Me_g, const float* __restrict__ rpW,
    const float* __restrict__ rp_b, float* __restrict__ rb_f32,
    bf16_t* __restrict__ rb_bf) {
  __shared__ float qrow[256], kq[256], readv[256];
  const int tid = threadIdx.x, b = blockIdx.x;
  qrow[tid] = qout[b * 256 + tid];
  __syncthreads();
  {
    const float* W = (tid < 128) ? semW : epiW;
    const int k = tid & 127;
    float acc = 0.f;
    for (int hh = 0; hh < 256; ++hh) acc = fmaf(W[k * 256 + hh], qrow[hh], acc);
    kq[tid] = acc;
  }
  __syncthreads();
  {
    const float* M = (tid < 128) ? Ms_g : Me_g;
    const float* q2 = (tid < 128) ? &kq[0] : &kq[128];
    const int idx = tid & 127;
    const float* row = M + (size_t)b * 16384 + (size_t)idx * 128;
    float acc = 0.f;
    for (int j = 0; j < 128; ++j) acc = fmaf(row[j], q2[j], acc);
    readv[tid] = acc;  // [cs | ce] = concat order
  }
  __syncthreads();
  {
    float acc = rp_b[tid];
    for (int hh = 0; hh < 256; ++hh)
      acc = fmaf(rpW[tid * 256 + hh], readv[hh], acc);
    rb_f32[b * 256 + tid] = acc;
    rb_bf[b * 256 + tid] = (bf16_t)acc;
  }
}

// ---------------------------------------------------------------------------
// K4: logits = rb(64x256) @ outW^T(256x32000) + out_b  via bf16 WMMA.
// 8 wave-tiles per block: 4 m-tiles x 2 n-tiles. 1000 blocks cover 4x2000.
// ---------------------------------------------------------------------------
__global__ void __launch_bounds__(256) out_gemm_kernel(
    const bf16_t* __restrict__ rb_bf, const bf16_t* __restrict__ outW,
    const float* __restrict__ out_b, float* __restrict__ out) {
  __shared__ bf16_t Ab[64][256];  // 32 KB, shared by all waves
  const int tid = threadIdx.x;
  for (int idx = tid; idx < 64 * 256; idx += 256)
    (&Ab[0][0])[idx] = rb_bf[idx];
  __syncthreads();

  const int w = tid >> 5, lane = tid & 31;
  const int tile = blockIdx.x * 8 + w;
  const int mt = tile & 3;      // 0..3  (rows 16*mt .. 16*mt+15)
  const int ntile = tile >> 2;  // 0..1999
  v8f acc = vzero8();
#pragma unroll
  for (int kt = 0; kt < 8; ++kt) {
    v16bf a = frag_a(&Ab[mt * 16][kt * 32], 256, lane);
    v16bf bb = frag_b(outW + (size_t)(ntile * 16) * 256 + kt * 32, 256, lane);
    acc = wmma_bf16(a, bb, acc);
  }
  const int n = lane & 15, hf = lane >> 4;
  const int col = ntile * 16 + n;
  const float bias = out_b[col];
#pragma unroll
  for (int r = 0; r < 8; ++r) {
    const int row = mt * 16 + r + 8 * hf;
    out[(size_t)row * 32000 + col] = acc[r] + bias;
  }
}

// ---------------------------------------------------------------------------
// launch
// ---------------------------------------------------------------------------
extern "C" void kernel_launch(void* const* d_in, const int* in_sizes, int n_in,
                              void* d_out, int out_size, void* d_ws,
                              size_t ws_size, hipStream_t stream) {
  const int* seq = (const int*)d_in[0];
  const float* embed_W = (const float*)d_in[1];
  const float* ff_w1 = (const float*)d_in[2];
  const float* ff_b1 = (const float*)d_in[3];
  const float* ff_w2 = (const float*)d_in[4];
  const float* ff_b2 = (const float*)d_in[5];
  const float* ln_g = (const float*)d_in[6];
  const float* ln_b = (const float*)d_in[7];
  const float* sem_W = (const float*)d_in[8];
  const float* epi_W = (const float*)d_in[9];
  const float* rp_W = (const float*)d_in[10];
  const float* rp_b = (const float*)d_in[11];
  const float* out_W = (const float*)d_in[12];
  const float* out_b = (const float*)d_in[13];
  float* out = (float*)d_out;

  char* ws = (char*)d_ws;
  size_t off = 0;
  auto alloc = [&](size_t bytes) {
    void* p = ws + off;
    off += (bytes + 255) & ~(size_t)255;
    return p;
  };
  bf16_t* w1_bf = (bf16_t*)alloc(512 * 256 * 2);
  bf16_t* w2_bf = (bf16_t*)alloc(256 * 512 * 2);
  bf16_t* sem_bf = (bf16_t*)alloc(128 * 256 * 2);
  bf16_t* epi_bf = (bf16_t*)alloc(128 * 256 * 2);
  bf16_t* outW_bf = (bf16_t*)alloc((size_t)32000 * 256 * 2);
  float* ks_all = (float*)alloc((size_t)2047 * 64 * 128 * 4);
  float* ke_all = (float*)alloc((size_t)2047 * 64 * 128 * 4);
  float* qout = (float*)alloc(64 * 256 * 4);
  float* Ms_out = (float*)alloc((size_t)64 * 128 * 128 * 4);
  float* Me_out = (float*)alloc((size_t)64 * 128 * 128 * 4);
  float* rb_f32 = (float*)alloc(64 * 256 * 4);
  bf16_t* rb_bf = (bf16_t*)alloc(64 * 256 * 2);

  // K0: weight conversions
  cvt_bf16_kernel<<<512, 256, 0, stream>>>(ff_w1, w1_bf, 512 * 256);
  cvt_bf16_kernel<<<512, 256, 0, stream>>>(ff_w2, w2_bf, 256 * 512);
  cvt_bf16_kernel<<<128, 256, 0, stream>>>(sem_W, sem_bf, 128 * 256);
  cvt_bf16_kernel<<<128, 256, 0, stream>>>(epi_W, epi_bf, 128 * 256);
  cvt_bf16_kernel<<<2048, 256, 0, stream>>>(out_W, outW_bf, 32000 * 256);

  // K1: fused token pipeline (131072 tokens / 64 per block)
  token_fused_kernel<<<2048, 128, 0, stream>>>(
      seq, embed_W, w1_bf, ff_b1, w2_bf, ff_b2, ln_g, ln_b, sem_bf, epi_bf,
      ks_all, ke_all, qout);

  // K2: sequential scan, one workgroup per batch
  scan_kernel<<<64, 256, 0, stream>>>(ks_all, ke_all, Ms_out, Me_out);

  // K3: readout
  readout_kernel<<<64, 256, 0, stream>>>(qout, sem_W, epi_W, Ms_out, Me_out,
                                         rp_W, rp_b, rb_f32, rb_bf);

  // K4: output GEMM (64 x 32000)
  out_gemm_kernel<<<1000, 256, 0, stream>>>(rb_bf, outW_bf, out_b, out);
}
